// HartleyMultiHeadAttention_84043920048617
// MI455X (gfx1250) — matrix operand order repaired and checked by reading, not compiled, not run on hardware
//
#include <hip/hip_runtime.h>
#include <hip/hip_bf16.h>

typedef _Float16 h16;
typedef __attribute__((ext_vector_type(16))) _Float16 v16h;
typedef __attribute__((ext_vector_type(8)))  _Float16 v8h;
typedef __attribute__((ext_vector_type(8)))  float    v8f;
typedef __attribute__((ext_vector_type(4)))  unsigned int u32x4;
typedef __attribute__((ext_vector_type(8)))  int i32x8;
typedef __attribute__((ext_vector_type(4)))  int i32x4;

#define WMMA(a,b,c) __builtin_amdgcn_wmma_f32_16x16x32_f16(false,(a),false,(b),(short)0,(c),false,false)

#define HAS_TDM __has_builtin(__builtin_amdgcn_tensor_load_to_lds)

#define LDA 40                      // padded LDS tile pitch (halfs): 80B, conflict-free, 16B aligned
#define G1_PLAIN 0x00010000u        // data_size=2B
#define G1_PAD   0x06D10000u        // data_size=2B | pad_enable | pad_interval=16dw | pad_amount=4dw

// ---------------- workspace layout (f16 element offsets) ----------------
#define MAT_CF   0u        // 32x64 forward cos
#define MAT_SF   2048u     // 32x64 forward sin
#define MAT_SFN  4096u     // -sin
#define MAT_PF   6144u     // cos+sin
#define MAT_QF   8192u     // sin-cos
#define MAT_GC   10240u    // 64x32 inverse cos  (scaled 1/64)
#define MAT_GS   12288u    // 64x32 inverse sin  (scaled 1/64)
#define MAT_GSN  14336u    // -inverse sin
#define MAT_PI   16384u    // (cos+sin)/64
#define MAT_QI   18432u    // (sin-cos)/64
#define MAT_WQT  20480u    // 128x64
#define MAT_WKT  28672u    // 128x64
#define MAT_WVT  36864u    // 128x64
#define MAT_WOT  45056u    // 16x128

#define OFF_R1   65536u                    // [b][d][kw][h][c] 2*64*32*64*64
#define OFF_I1   (OFF_R1 + 16777216u)
#define OFF_QT   OFF_R1                    // alias (r1/i1 dead): [bh][tok][feat]
#define OFF_KT   (OFF_QT + 8388608u)
#define OFF_VT   (OFF_KT + 8388608u)       // [bh][feat][tok]
#define OFF_R2   (OFF_R1 + 33554432u)      // [b][kw][kh][d][c] 2*32*32*64*64
#define OFF_I2   (OFF_R2 + 8388608u)
#define OFF_ATT  OFF_R2                    // alias: [bh][q][k] 16*512*512
#define OFF_OT   (OFF_R2 + 4194304u)       // [bh][q][feat] 16*512*1024
#define OFF_XT   (OFF_R2 + 16777216u)      // [b][a2][a1][a0][c] 2*32^3*64
#define OFF_OP   OFF_XT                    // alias (xt dead): [b][a2][a1][a0][o16]
#define OFF_RA   (OFF_XT + 1048576u)       // [b][a2][d][a1][o] 2*32*64*32*16
#define OFF_IA   (OFF_RA + 2097152u)
#define OFF_RB   (OFF_IA + 2097152u)       // [b][d][h][a2][o] 2*64*64*32*16
#define OFF_IB   (OFF_RB + 4194304u)

// ---------------- fragment helpers (wave32) ----------------
__device__ __forceinline__ v8f vzero() {
  v8f v;
#pragma unroll
  for (int i = 0; i < 8; i++) v[i] = 0.0f;
  return v;
}

// A fragment from row-major f16 [M][K] (ld elems): rows m0..m0+15, K chunk k0..k0+31
__device__ __forceinline__ v16h load_a_rm(const h16* p, int m0, int k0, int ld) {
  int lane = threadIdx.x & 31;
  int r = lane & 15, hl = lane >> 4;
  const h16* q = p + (size_t)(m0 + r) * ld + k0 + hl * 16;
  v8h lo = *(const v8h*)q;
  v8h hi = *(const v8h*)(q + 8);
  v16h o;
#pragma unroll
  for (int i = 0; i < 8; i++) { o[i] = lo[i]; o[8 + i] = hi[i]; }
  return o;
}

// B fragment from NT storage: row-major f16 [N][K] (B^T), cols n0..+15, K k0..+31
__device__ __forceinline__ v16h load_b_nt(const h16* p, int n0, int k0, int ld) {
  int lane = threadIdx.x & 31;
  int n = lane & 15, hl = lane >> 4;
  const h16* q = p + (size_t)(n0 + n) * ld + k0 + hl * 16;
  v8h lo = *(const v8h*)q;
  v8h hi = *(const v8h*)(q + 8);
  v16h o;
#pragma unroll
  for (int i = 0; i < 8; i++) { o[i] = lo[i]; o[8 + i] = hi[i]; }
  return o;
}

// B fragment gathered from K-major f16 [K][N] (k stride = ldk elems) - small tiles only
__device__ __forceinline__ v16h load_b_km_f16(const h16* p, int n0, int k0, int ldk) {
  int lane = threadIdx.x & 31;
  int n = n0 + (lane & 15), hl = lane >> 4;
  v16h o;
#pragma unroll
  for (int i = 0; i < 16; i++) o[i] = p[(size_t)(k0 + hl * 16 + i) * ldk + n];
  return o;
}

__device__ __forceinline__ float selu_f(float x) {
  const float l = 1.0507009873554805f, a = 1.6732632423543772f;
  return x > 0.f ? l * x : l * a * (__expf(x) - 1.f);
}

__device__ __forceinline__ unsigned lds_off(const void* p) {
  // generic pointer to LDS: low 32 bits are the wave-relative LDS byte address
  return (unsigned)(uintptr_t)p;
}

// Issue one TDM 2D tile op: th rows x tw f16 elems, global row stride ld (elems).
__device__ __forceinline__ void tdm_issue(const h16* gptr, unsigned ldsb, int ld,
                                          int tw, int th, unsigned w0, bool store) {
#if HAS_TDM
  unsigned long long ga = (unsigned long long)(uintptr_t)gptr;
  u32x4 g0;
  g0[0] = 1u;                                                 // count=1, user mode
  g0[1] = ldsb;                                               // lds_addr
  g0[2] = (unsigned)ga;                                       // global_addr[31:0]
  g0[3] = (unsigned)((ga >> 32) & 0x1FFFFFFull) | (2u << 30); // addr[56:32] | type=2
  i32x8 g1;
  g1[0] = (int)w0;                                            // data_size | pad config
  g1[1] = (int)(((unsigned)ld & 0xFFFFu) << 16);              // tensor_dim0 lo16
  g1[2] = (int)((unsigned)ld >> 16);                          // dim0 hi16 | dim1 lo16 (0)
  g1[3] = (int)(1u | ((unsigned)tw << 16));                   // dim1 hi16 (dim1=65536), tile_dim0
  g1[4] = th;                                                 // tile_dim1
  g1[5] = ld;                                                 // tensor_dim0_stride lo32
  g1[6] = 0; g1[7] = 0;
  i32x4 z4; z4[0] = 0; z4[1] = 0; z4[2] = 0; z4[3] = 0;
#if defined(__clang_major__) && __clang_major__ >= 23
  i32x8 z8;
#pragma unroll
  for (int i = 0; i < 8; i++) z8[i] = 0;
  if (store) __builtin_amdgcn_tensor_store_from_lds(g0, g1, z4, z4, z8, 0);
  else       __builtin_amdgcn_tensor_load_to_lds(g0, g1, z4, z4, z8, 0);
#else
  if (store) __builtin_amdgcn_tensor_store_from_lds(g0, g1, z4, z4, 0);
  else       __builtin_amdgcn_tensor_load_to_lds(g0, g1, z4, z4, 0);
#endif
#else
  (void)gptr; (void)ldsb; (void)ld; (void)tw; (void)th; (void)w0; (void)store;
#endif
}

// fallback cooperative tile copy (256 threads): 128x32 f16, dst pitch LDA
__device__ __forceinline__ void coop_tile_128x32(h16* dst, const h16* src, int ld) {
  int t = threadIdx.x;
  int r = t >> 1, half = t & 1;
  const h16* s = src + (size_t)r * ld + half * 16;
  h16* d = dst + r * LDA + half * 16;
  *(v8h*)d = *(const v8h*)s;
  *(v8h*)(d + 8) = *(const v8h*)(s + 8);
}

// ---------------- kernel 0: tables + weight transposes ----------------
__global__ void k_precompute(const float* __restrict__ Wq, const float* __restrict__ Wk,
                             const float* __restrict__ Wv, const float* __restrict__ Wo,
                             h16* __restrict__ W) {
  int t = blockIdx.x * blockDim.x + threadIdx.x;       // 8192 threads
  const float STEP = 6.28318530717958647692f / 64.f;
  if (t < 2048) {
    int j = t >> 6, n = t & 63;
    int k = (j < 16) ? j : j + 32;
    int ph = (k * n) & 63;
    float c = __cosf(ph * STEP), s = __sinf(ph * STEP);
    W[MAT_CF + t] = (h16)c;   W[MAT_SF + t] = (h16)s;  W[MAT_SFN + t] = (h16)(-s);
    W[MAT_PF + t] = (h16)(c + s); W[MAT_QF + t] = (h16)(s - c);
    int d = t >> 5, j2 = t & 31;
    int k2 = (j2 < 16) ? j2 : j2 + 32;
    int ph2 = (d * k2) & 63;
    float c2 = __cosf(ph2 * STEP), s2 = __sinf(ph2 * STEP);
    const float sc = 1.0f / 64.f;    // 1/64 per stage -> 1/N over 3 stages
    W[MAT_GC + t] = (h16)(sc * c2); W[MAT_GS + t] = (h16)(sc * s2);
    W[MAT_GSN + t] = (h16)(-sc * s2);
    W[MAT_PI + t] = (h16)(sc * (c2 + s2)); W[MAT_QI + t] = (h16)(sc * (s2 - c2));
    int o = t >> 7, i = t & 127;
    W[MAT_WOT + t] = (h16)Wo[i * 16 + o];
  }
  int n = t >> 6, c = t & 63;
  W[MAT_WQT + t] = (h16)Wq[c * 128 + n];
  W[MAT_WKT + t] = (h16)Wk[c * 128 + n];
  W[MAT_WVT + t] = (h16)Wv[c * 128 + n];
}

// ---------------- forward DHT stage 1 (W axis): x(b,d,h,W,C) -> r1,i1 ----------------
__global__ __launch_bounds__(256) void k_dht_w(const float* __restrict__ x, h16* __restrict__ W) {
  __shared__ h16 T[2][64][72];                 // transposed slab [c][w], padded
  int tid = threadIdx.x;
  int wave = tid >> 5;
  int slabL = wave >> 2;
  int ntile = wave & 3;
  int slab0 = blockIdx.x * 2;
  {
    int g = tid >> 7, t = tid & 127;           // 128 threads per slab
    const float* xs = x + (size_t)(slab0 + g) * 4096;
#pragma unroll
    for (int q = 0; q < 8; q++) {
      int idx = (q * 128 + t) * 4;             // coalesced float4
      float4 v = *(const float4*)(xs + idx);
      int w = idx >> 6, c = idx & 63;
      T[g][c + 0][w] = (h16)v.x; T[g][c + 1][w] = (h16)v.y;
      T[g][c + 2][w] = (h16)v.z; T[g][c + 3][w] = (h16)v.w;
    }
  }
  __syncthreads();
  v8f aC[2] = {vzero(), vzero()}, aS[2] = {vzero(), vzero()};
#pragma unroll
  for (int ks = 0; ks < 2; ks++) {
    int k0 = ks * 32;
    v16h bf = load_b_nt(&T[slabL][0][0], ntile * 16, k0, 72);
#pragma unroll
    for (int mt = 0; mt < 2; mt++) {
      v16h ac = load_a_rm(W + MAT_CF, mt * 16, k0, 64);
      v16h as = load_a_rm(W + MAT_SFN, mt * 16, k0, 64);   // i = -(S @ x)
      aC[mt] = WMMA(ac, bf, aC[mt]);
      aS[mt] = WMMA(as, bf, aS[mt]);
    }
  }
  int slab = slab0 + slabL;
  int h = slab & 63, bd = slab >> 6;
  int lane = tid & 31, hl = lane >> 4;
  int c = ntile * 16 + (lane & 15);
  h16* r1 = W + OFF_R1; h16* i1 = W + OFF_I1;
#pragma unroll
  for (int mt = 0; mt < 2; mt++)
#pragma unroll
    for (int j = 0; j < 8; j++) {
      int kw = mt * 16 + hl * 8 + j;
      size_t o = (((size_t)bd * 32 + kw) * 64 + h) * 64 + c;
      r1[o] = (h16)aC[mt][j];
      i1[o] = (h16)aS[mt][j];
    }
}

// cooperative transposed staging of 2 complex f16 slabs (64x64 each)
__device__ __forceinline__ void stage2_slabs(h16 Tr[2][64][72], h16 Ti[2][64][72],
                                             const h16* rbase, const h16* ibase,
                                             size_t s0) {
  int tid = threadIdx.x;
#pragma unroll
  for (int q = 0; q < 8; q++) {
    int u = (q * 256 + tid) * 8;               // 16384 halfs total
    int g = u >> 13, comp = (u >> 12) & 1, e = u & 4095;
    const h16* src = (comp ? ibase : rbase) + (s0 + g) * 4096 + e;
    v8h v = *(const v8h*)src;
    int k = e >> 6, c0 = e & 63;
#pragma unroll
    for (int i = 0; i < 8; i++) {
      if (comp) Ti[g][c0 + i][k] = v[i];
      else      Tr[g][c0 + i][k] = v[i];
    }
  }
}

// ---------------- forward DHT stage 2 (H axis): r1,i1 -> r2,i2 ----------------
__global__ __launch_bounds__(256) void k_dht_h(h16* __restrict__ W) {
  __shared__ h16 Tr[2][64][72], Ti[2][64][72];
  int tid = threadIdx.x;
  int wave = tid >> 5;
  int slabL = wave >> 2, ntile = wave & 3;
  size_t s0 = (size_t)blockIdx.x * 2;
  stage2_slabs(Tr, Ti, W + OFF_R1, W + OFF_I1, s0);
  __syncthreads();
  v8f ar[2] = {vzero(), vzero()}, ai[2] = {vzero(), vzero()};
#pragma unroll
  for (int ks = 0; ks < 2; ks++) {
    int k0 = ks * 32;
    v16h br = load_b_nt(&Tr[slabL][0][0], ntile * 16, k0, 72);
    v16h bi = load_b_nt(&Ti[slabL][0][0], ntile * 16, k0, 72);
#pragma unroll
    for (int mt = 0; mt < 2; mt++) {
      v16h ac = load_a_rm(W + MAT_CF, mt * 16, k0, 64);
      v16h as = load_a_rm(W + MAT_SF, mt * 16, k0, 64);
      v16h an = load_a_rm(W + MAT_SFN, mt * 16, k0, 64);
      ar[mt] = WMMA(ac, br, ar[mt]);  ar[mt] = WMMA(as, bi, ar[mt]);  // r' = C r + S i
      ai[mt] = WMMA(ac, bi, ai[mt]);  ai[mt] = WMMA(an, br, ai[mt]);  // i' = C i - S r
    }
  }
  int s = (int)s0 + slabL;                     // (b*64+d)*32+kw
  int b = s >> 11, d = (s >> 5) & 63, kw = s & 31;
  int lane = tid & 31, hl = lane >> 4;
  int c = ntile * 16 + (lane & 15);
  h16* r2 = W + OFF_R2; h16* i2 = W + OFF_I2;
#pragma unroll
  for (int mt = 0; mt < 2; mt++)
#pragma unroll
    for (int j = 0; j < 8; j++) {
      int kh = mt * 16 + hl * 8 + j;
      size_t o = ((((size_t)b * 32 + kw) * 32 + kh) * 64 + d) * 64 + c;
      r2[o] = (h16)ar[mt][j];
      i2[o] = (h16)ai[mt][j];
    }
}

// ---------------- forward DHT stage 3 (D axis, fold r-i combine): -> xt ----------------
__global__ __launch_bounds__(256) void k_dht_d(h16* __restrict__ W) {
  __shared__ h16 Tr[2][64][72], Ti[2][64][72];
  int tid = threadIdx.x;
  int wave = tid >> 5;
  int slabL = wave >> 2, ntile = wave & 3;
  size_t s0 = (size_t)blockIdx.x * 2;
  stage2_slabs(Tr, Ti, W + OFF_R2, W + OFF_I2, s0);
  __syncthreads();
  v8f acc[2] = {vzero(), vzero()};
#pragma unroll
  for (int ks = 0; ks < 2; ks++) {
    int k0 = ks * 32;
    v16h br = load_b_nt(&Tr[slabL][0][0], ntile * 16, k0, 72);
    v16h bi = load_b_nt(&Ti[slabL][0][0], ntile * 16, k0, 72);
#pragma unroll
    for (int mt = 0; mt < 2; mt++) {
      v16h ap = load_a_rm(W + MAT_PF, mt * 16, k0, 64);    // (C+S)
      v16h aq = load_a_rm(W + MAT_QF, mt * 16, k0, 64);    // (S-C)
      acc[mt] = WMMA(ap, br, acc[mt]);
      acc[mt] = WMMA(aq, bi, acc[mt]);
    }
  }
  int s = (int)s0 + slabL;                     // (b*32+kw)*32+kh
  int lane = tid & 31, hl = lane >> 4;
  int c = ntile * 16 + (lane & 15);
  h16* xt = W + OFF_XT;
#pragma unroll
  for (int mt = 0; mt < 2; mt++)
#pragma unroll
    for (int j = 0; j < 8; j++) {
      int kd = mt * 16 + hl * 8 + j;
      xt[(size_t)s * 2048 + (size_t)kd * 64 + c] = (h16)acc[mt][j];
    }
}

// ---------------- Q/K/V projection + patch-group scatter ----------------
__global__ __launch_bounds__(256) void k_proj(h16* __restrict__ W) {
  int wave = threadIdx.x >> 5;
  int wy = wave >> 2, wx = wave & 3;
  int m0 = blockIdx.x * 128 + wy * 64;
  int n0 = wx * 32;
  const h16* xt = W + OFF_XT;
  const h16* Bw = W + (blockIdx.y == 0 ? MAT_WQT : blockIdx.y == 1 ? MAT_WKT : MAT_WVT);
  v8f acc[4][2];
#pragma unroll
  for (int mt = 0; mt < 4; mt++)
#pragma unroll
    for (int nt = 0; nt < 2; nt++) acc[mt][nt] = vzero();
#pragma unroll
  for (int ks = 0; ks < 2; ks++) {
    int k0 = ks * 32;
    v16h bf[2];
#pragma unroll
    for (int nt = 0; nt < 2; nt++) bf[nt] = load_b_nt(Bw, n0 + nt * 16, k0, 64);
#pragma unroll
    for (int mt = 0; mt < 4; mt++) {
      v16h af = load_a_rm(xt, m0 + mt * 16, k0, 64);
#pragma unroll
      for (int nt = 0; nt < 2; nt++) acc[mt][nt] = WMMA(af, bf[nt], acc[mt][nt]);
    }
  }
  int lane = threadIdx.x & 31, hl = lane >> 4;
  int p = blockIdx.y;
  h16* dst = W + (p == 0 ? OFF_QT : p == 1 ? OFF_KT : OFF_VT);
#pragma unroll
  for (int mt = 0; mt < 4; mt++)
#pragma unroll
    for (int nt = 0; nt < 2; nt++)
#pragma unroll
      for (int j = 0; j < 8; j++) {
        int m = m0 + mt * 16 + hl * 8 + j;
        int n = n0 + nt * 16 + (lane & 15);
        int b = m >> 15, a = m & 32767;
        int a2 = a >> 10, a1 = (a >> 5) & 31, a0 = a & 31;
        int tok = ((a0 >> 2) * 8 + (a1 >> 2)) * 8 + (a2 >> 2);
        int kap = n >> 3, hh = n & 7;
        int feat = ((kap * 4 + (a0 & 3)) * 4 + (a1 & 3)) * 4 + (a2 & 3);
        int bh = b * 8 + hh;
        size_t off = (p == 2) ? ((size_t)bh * 1024 + feat) * 512 + tok     // Vt [bh][feat][tok]
                              : ((size_t)bh * 512 + tok) * 1024 + feat;    // Qt/Kt [bh][tok][feat]
        dst[off] = (h16)acc[mt][nt][j];
      }
}

// ---------------- attention GEMM 1: attn = selu(Q K^T), per (b,h) 512x512x1024 ----------------
__global__ __launch_bounds__(256) void k_attn_qk(h16* __restrict__ W) {
  __shared__ __attribute__((aligned(16))) h16 smem[20480];  // 2x(A,B) tiles pitch 40; reused as 128x128 out
  int wave = threadIdx.x >> 5;
  int wy = wave >> 2, wx = wave & 3;
  int bh = blockIdx.y;
  int rowA = (blockIdx.x >> 2) * 128;
  int rowB = (blockIdx.x & 3) * 128;
  const h16* Q = W + OFF_QT + (size_t)bh * 524288;
  const h16* K = W + OFF_KT + (size_t)bh * 524288;
  v8f acc[4][2];
#pragma unroll
  for (int mt = 0; mt < 4; mt++)
#pragma unroll
    for (int nt = 0; nt < 2; nt++) acc[mt][nt] = vzero();
  int lane = threadIdx.x & 31, hl = lane >> 4;
#if HAS_TDM
  if (wave == 0) tdm_issue(Q + (size_t)rowA * 1024, lds_off(smem), 1024, 32, 128, G1_PAD, false);
  if (wave == 1) tdm_issue(K + (size_t)rowB * 1024, lds_off(smem + 10240), 1024, 32, 128, G1_PAD, false);
  if (wave < 2) __builtin_amdgcn_s_wait_tensorcnt(0);
  __syncthreads();
  for (int ks = 0; ks < 32; ks++) {
    int cur = ks & 1, nxt = cur ^ 1;
    if (ks + 1 < 32) {
      if (wave == 0) tdm_issue(Q + (size_t)rowA * 1024 + (ks + 1) * 32, lds_off(smem + nxt * 5120), 1024, 32, 128, G1_PAD, false);
      if (wave == 1) tdm_issue(K + (size_t)rowB * 1024 + (ks + 1) * 32, lds_off(smem + 10240 + nxt * 5120), 1024, 32, 128, G1_PAD, false);
    }
    h16* curA = smem + cur * 5120;
    h16* curB = smem + 10240 + cur * 5120;
    v16h bf[2];
#pragma unroll
    for (int nt = 0; nt < 2; nt++) bf[nt] = load_b_nt(curB, wx * 32 + nt * 16, 0, LDA);
#pragma unroll
    for (int mt = 0; mt < 4; mt++) {
      v16h af = load_a_rm(curA, wy * 64 + mt * 16, 0, LDA);
#pragma unroll
      for (int nt = 0; nt < 2; nt++) acc[mt][nt] = WMMA(af, bf[nt], acc[mt][nt]);
    }
    if (wave < 2 && ks + 1 < 32) __builtin_amdgcn_s_wait_tensorcnt(0);
    __syncthreads();
  }
  // epilogue: SELU, pack tile in LDS, one TDM store DMA
#pragma unroll
  for (int mt = 0; mt < 4; mt++)
#pragma unroll
    for (int nt = 0; nt < 2; nt++)
#pragma unroll
      for (int j = 0; j < 8; j++) {
        int rl = wy * 64 + mt * 16 + hl * 8 + j;
        int cl = wx * 32 + nt * 16 + (lane & 15);
        smem[rl * 128 + cl] = (h16)selu_f(acc[mt][nt][j]);   // 128x128 overlay
      }
  __syncthreads();
  h16* att = W + OFF_ATT + (size_t)bh * 262144;
  if (wave == 0) {
    tdm_issue(att + (size_t)rowA * 512 + rowB, lds_off(smem), 512, 128, 128, G1_PLAIN, true);
    __builtin_amdgcn_s_wait_tensorcnt(0);
  }
#else
  for (int ks = 0; ks < 32; ks++) {
    coop_tile_128x32(smem, Q + (size_t)rowA * 1024 + ks * 32, 1024);
    coop_tile_128x32(smem + 10240, K + (size_t)rowB * 1024 + ks * 32, 1024);
    __syncthreads();
    v16h bf[2];
#pragma unroll
    for (int nt = 0; nt < 2; nt++) bf[nt] = load_b_nt(smem + 10240, wx * 32 + nt * 16, 0, LDA);
#pragma unroll
    for (int mt = 0; mt < 4; mt++) {
      v16h af = load_a_rm(smem, wy * 64 + mt * 16, 0, LDA);
#pragma unroll
      for (int nt = 0; nt < 2; nt++) acc[mt][nt] = WMMA(af, bf[nt], acc[mt][nt]);
    }
    __syncthreads();
  }
  h16* att = W + OFF_ATT + (size_t)bh * 262144;
#pragma unroll
  for (int mt = 0; mt < 4; mt++)
#pragma unroll
    for (int nt = 0; nt < 2; nt++)
#pragma unroll
      for (int j = 0; j < 8; j++) {
        int q = rowA + wy * 64 + mt * 16 + hl * 8 + j;
        int kc = rowB + wx * 32 + nt * 16 + (lane & 15);
        att[(size_t)q * 512 + kc] = (h16)selu_f(acc[mt][nt][j]);
      }
#endif
}

// ---------------- attention GEMM 2: Ot = attn @ V^T, per (b,h) 512x1024x512 ----------------
__global__ __launch_bounds__(256) void k_attn_av(h16* __restrict__ W) {
  __shared__ __attribute__((aligned(16))) h16 smem[20480];
  int wave = threadIdx.x >> 5;
  int wy = wave >> 2, wx = wave & 3;
  int bh = blockIdx.y;
  int rowA = (blockIdx.x >> 3) * 128;
  int rowB = (blockIdx.x & 7) * 128;
  const h16* A = W + OFF_ATT + (size_t)bh * 262144;   // [q][k] ld 512
  const h16* V = W + OFF_VT + (size_t)bh * 524288;    // [feat][tok] ld 512
  v8f acc[4][2];
#pragma unroll
  for (int mt = 0; mt < 4; mt++)
#pragma unroll
    for (int nt = 0; nt < 2; nt++) acc[mt][nt] = vzero();
  int lane = threadIdx.x & 31, hl = lane >> 4;
#if HAS_TDM
  if (wave == 0) tdm_issue(A + (size_t)rowA * 512, lds_off(smem), 512, 32, 128, G1_PAD, false);
  if (wave == 1) tdm_issue(V + (size_t)rowB * 512, lds_off(smem + 10240), 512, 32, 128, G1_PAD, false);
  if (wave < 2) __builtin_amdgcn_s_wait_tensorcnt(0);
  __syncthreads();
  for (int ks = 0; ks < 16; ks++) {
    int cur = ks & 1, nxt = cur ^ 1;
    if (ks + 1 < 16) {
      if (wave == 0) tdm_issue(A + (size_t)rowA * 512 + (ks + 1) * 32, lds_off(smem + nxt * 5120), 512, 32, 128, G1_PAD, false);
      if (wave == 1) tdm_issue(V + (size_t)rowB * 512 + (ks + 1) * 32, lds_off(smem + 10240 + nxt * 5120), 512, 32, 128, G1_PAD, false);
    }
    h16* curA = smem + cur * 5120;
    h16* curB = smem + 10240 + cur * 5120;
    v16h bf[2];
#pragma unroll
    for (int nt = 0; nt < 2; nt++) bf[nt] = load_b_nt(curB, wx * 32 + nt * 16, 0, LDA);
#pragma unroll
    for (int mt = 0; mt < 4; mt++) {
      v16h af = load_a_rm(curA, wy * 64 + mt * 16, 0, LDA);
#pragma unroll
      for (int nt = 0; nt < 2; nt++) acc[mt][nt] = WMMA(af, bf[nt], acc[mt][nt]);
    }
    if (wave < 2 && ks + 1 < 16) __builtin_amdgcn_s_wait_tensorcnt(0);
    __syncthreads();
  }
#pragma unroll
  for (int mt = 0; mt < 4; mt++)
#pragma unroll
    for (int nt = 0; nt < 2; nt++)
#pragma unroll
      for (int j = 0; j < 8; j++) {
        int rl = wy * 64 + mt * 16 + hl * 8 + j;
        int cl = wx * 32 + nt * 16 + (lane & 15);
        smem[rl * 128 + cl] = (h16)acc[mt][nt][j];          // 128x128 overlay
      }
  __syncthreads();
  h16* Ot = W + OFF_OT + (size_t)bh * 524288;
  if (wave == 0) {
    tdm_issue(Ot + (size_t)rowA * 1024 + rowB, lds_off(smem), 1024, 128, 128, G1_PLAIN, true);
    __builtin_amdgcn_s_wait_tensorcnt(0);
  }
#else
  for (int ks = 0; ks < 16; ks++) {
    coop_tile_128x32(smem, A + (size_t)rowA * 512 + ks * 32, 512);
    coop_tile_128x32(smem + 10240, V + (size_t)rowB * 512 + ks * 32, 512);
    __syncthreads();
    v16h bf[2];
#pragma unroll
    for (int nt = 0; nt < 2; nt++) bf[nt] = load_b_nt(smem + 10240, wx * 32 + nt * 16, 0, LDA);
#pragma unroll
    for (int mt = 0; mt < 4; mt++) {
      v16h af = load_a_rm(smem, wy * 64 + mt * 16, 0, LDA);
#pragma unroll
      for (int nt = 0; nt < 2; nt++) acc[mt][nt] = WMMA(af, bf[nt], acc[mt][nt]);
    }
    __syncthreads();
  }
  h16* Ot = W + OFF_OT + (size_t)bh * 524288;
#pragma unroll
  for (int mt = 0; mt < 4; mt++)
#pragma unroll
    for (int nt = 0; nt < 2; nt++)
#pragma unroll
      for (int j = 0; j < 8; j++) {
        int q = rowA + wy * 64 + mt * 16 + hl * 8 + j;
        int c = rowB + wx * 32 + nt * 16 + (lane & 15);
        Ot[(size_t)q * 1024 + c] = (h16)acc[mt][nt][j];
      }
#endif
}

// ---------------- output projection: op[b][a][o16] = ungroup(Ot) @ Wo ----------------
__global__ __launch_bounds__(256) void k_wo(h16* __restrict__ W) {
  int wave = threadIdx.x >> 5;
  int m0 = blockIdx.x * 512 + wave * 64;
  int lane = threadIdx.x & 31;
  int r = lane & 15, hl = lane >> 4;
  const h16* Ot = W + OFF_OT;
  const h16* Bw = W + MAT_WOT;                 // [o(16)][i(128)]
  v8f acc[4] = {vzero(), vzero(), vzero(), vzero()};
#pragma unroll
  for (int ks = 0; ks < 4; ks++) {
    int k0 = ks * 32;
    v16h bf = load_b_nt(Bw, 0, k0, 128);
#pragma unroll
    for (int mt = 0; mt < 4; mt++) {
      int m = m0 + mt * 16 + r;
      int b = m >> 15, a = m & 32767;
      int a2 = a >> 10, a1 = (a >> 5) & 31, a0 = a & 31;
      int tok = ((a0 >> 2) * 8 + (a1 >> 2)) * 8 + (a2 >> 2);
      int pd = a0 & 3, ph = a1 & 3, pw = a2 & 3;
      v16h af;
#pragma unroll
      for (int half = 0; half < 2; half++) {
        int kc = k0 + hl * 16 + half * 8;          // fixed v, head 0..7
        int v = kc >> 3;
        int feat = ((v * 4 + pd) * 4 + ph) * 4 + pw;
        size_t base = ((size_t)(b * 8) * 512 + tok) * 1024 + feat;
#pragma unroll
        for (int i = 0; i < 8; i++) af[half * 8 + i] = Ot[base + (size_t)i * 524288];
      }
      acc[mt] = WMMA(af, bf, acc[mt]);
    }
  }
  h16* op = W + OFF_OP;
#pragma unroll
  for (int mt = 0; mt < 4; mt++)
#pragma unroll
    for (int j = 0; j < 8; j++) {
      int m = m0 + mt * 16 + hl * 8 + j;
      op[(size_t)m * 16 + (lane & 15)] = (h16)acc[mt][j];
    }
}

// ---------------- inverse DHT stage A (D axis, 32 kept -> 64) ----------------
__global__ __launch_bounds__(256) void k_inv_d(h16* __restrict__ W) {
  int job = blockIdx.x * 8 + (threadIdx.x >> 5);  // [0,8192)
  int s = job >> 2, mt = job & 3;                 // s = (b*32+a2)*32+a1
  const h16* ops = W + OFF_OP + (size_t)s * 512;  // 32(k) x 16(o)
  v16h bf = load_b_km_f16(ops, 0, 0, 16);
  v16h ac = load_a_rm(W + MAT_GC, mt * 16, 0, 32);
  v16h an = load_a_rm(W + MAT_GSN, mt * 16, 0, 32);
  v8f aR = vzero(), aI = vzero();
  aR = WMMA(ac, bf, aR);
  aI = WMMA(an, bf, aI);
  int b = s >> 10, a2 = (s >> 5) & 31, a1 = s & 31;
  int lane = threadIdx.x & 31, hl = lane >> 4, o = lane & 15;
  h16* rA = W + OFF_RA; h16* iA = W + OFF_IA;
#pragma unroll
  for (int j = 0; j < 8; j++) {
    int d = mt * 16 + hl * 8 + j;
    size_t off = ((((size_t)b * 32 + a2) * 64 + d) * 32 + a1) * 16 + o;
    rA[off] = (h16)aR[j];
    iA[off] = (h16)aI[j];
  }
}

// ---------------- inverse DHT stage B (H axis) ----------------
__global__ __launch_bounds__(256) void k_inv_h(h16* __restrict__ W) {
  int job = blockIdx.x * 8 + (threadIdx.x >> 5);  // [0,16384)
  int s = job >> 2, mt = job & 3;                 // s = (b*32+a2)*64+d
  const h16* rs = W + OFF_RA + (size_t)s * 512;
  const h16* is = W + OFF_IA + (size_t)s * 512;
  v16h br = load_b_km_f16(rs, 0, 0, 16);
  v16h bi = load_b_km_f16(is, 0, 0, 16);
  v16h ac = load_a_rm(W + MAT_GC, mt * 16, 0, 32);
  v16h as = load_a_rm(W + MAT_GS, mt * 16, 0, 32);
  v16h an = load_a_rm(W + MAT_GSN, mt * 16, 0, 32);
  v8f aR = vzero(), aI = vzero();
  aR = WMMA(ac, br, aR);  aR = WMMA(as, bi, aR);
  aI = WMMA(ac, bi, aI);  aI = WMMA(an, br, aI);
  int b = s >> 11, a2 = (s >> 6) & 31, d = s & 63;
  int lane = threadIdx.x & 31, hl = lane >> 4, o = lane & 15;
  h16* rB = W + OFF_RB; h16* iB = W + OFF_IB;
#pragma unroll
  for (int j = 0; j < 8; j++) {
    int h = mt * 16 + hl * 8 + j;
    size_t off = ((((size_t)b * 64 + d) * 64 + h) * 32 + a2) * 16 + o;
    rB[off] = (h16)aR[j];
    iB[off] = (h16)aI[j];
  }
}

// ---------------- inverse DHT stage C (W axis, fold r-i combine) -> y f32 ----------------
__global__ __launch_bounds__(256) void k_inv_w(h16* __restrict__ W, float* __restrict__ y) {
  int job = blockIdx.x * 8 + (threadIdx.x >> 5);  // [0,32768)
  int s = job >> 2, mt = job & 3;                 // s = (b*64+d)*64+h
  const h16* rs = W + OFF_RB + (size_t)s * 512;
  const h16* is = W + OFF_IB + (size_t)s * 512;
  v16h br = load_b_km_f16(rs, 0, 0, 16);
  v16h bi = load_b_km_f16(is, 0, 0, 16);
  v16h ap = load_a_rm(W + MAT_PI, mt * 16, 0, 32);
  v16h aq = load_a_rm(W + MAT_QI, mt * 16, 0, 32);
  v8f acc = vzero();
  acc = WMMA(ap, br, acc);
  acc = WMMA(aq, bi, acc);
  int lane = threadIdx.x & 31, hl = lane >> 4, o = lane & 15;
#pragma unroll
  for (int j = 0; j < 8; j++) {
    int w = mt * 16 + hl * 8 + j;
    y[((size_t)s * 64 + w) * 16 + o] = acc[j];    // (B,D,H,W,16) contiguous
  }
}

// ---------------- launcher ----------------
extern "C" void kernel_launch(void* const* d_in, const int* in_sizes, int n_in,
                              void* d_out, int out_size, void* d_ws, size_t ws_size,
                              hipStream_t stream) {
  (void)in_sizes; (void)n_in; (void)out_size; (void)ws_size;
  const float* x  = (const float*)d_in[0];
  const float* Wq = (const float*)d_in[1];
  const float* Wk = (const float*)d_in[2];
  const float* Wv = (const float*)d_in[3];
  const float* Wo = (const float*)d_in[4];
  float* y = (float*)d_out;
  h16* W = (h16*)d_ws;

  k_precompute<<<32, 256, 0, stream>>>(Wq, Wk, Wv, Wo, W);
  k_dht_w<<<4096, 256, 0, stream>>>(x, W);
  k_dht_h<<<2048, 256, 0, stream>>>(W);
  k_dht_d<<<1024, 256, 0, stream>>>(W);
  k_proj<<<dim3(512, 3), 256, 0, stream>>>(W);
  k_attn_qk<<<dim3(16, 16), 256, 0, stream>>>(W);
  k_attn_av<<<dim3(32, 16), 256, 0, stream>>>(W);
  k_wo<<<128, 256, 0, stream>>>(W);
  k_inv_d<<<1024, 256, 0, stream>>>(W);
  k_inv_h<<<2048, 256, 0, stream>>>(W);
  k_inv_w<<<4096, 256, 0, stream>>>(W, y);
}
